// EnhancedCricketGNN_41832981463312
// MI455X (gfx1250) — matrix operand order repaired and checked by reading, not compiled, not run on hardware
//
#include <hip/hip_runtime.h>
#include <cfloat>
#include <cstdint>

// ---------------------------------------------------------------------------
// EnhancedCricketGNN forward for gfx1250 (MI455X).
// Dense matmuls -> v_wmma_f32_16x16x32_bf16 (f32 accumulate), 32x32 macro-tile
// per wave (2x2 register blocking: 8 b128 fragment loads feed 4 WMMAs).
// Edge softmax/scatter -> wave32 lane-per-channel + f32 atomics (HBM bound).
// ---------------------------------------------------------------------------

typedef __attribute__((ext_vector_type(16))) __bf16 v16bf;
typedef __attribute__((ext_vector_type(8)))  __bf16 v8bf;
typedef __attribute__((ext_vector_type(8)))  float  v8f;

static constexpr int NP  = 100000;
static constexpr int NV  = 50000;
static constexpr int HD  = 256;   // hidden dim
static constexpr int NH  = 8;     // heads
static constexpr int NE  = 200000;
static constexpr int LYR = 3;

// ---- bf16 <-> f32 via bit ops (storage-only use of __bf16) ----------------
__device__ inline __bf16 f2bf(float f) {
  unsigned u = __float_as_uint(f);
  unsigned r = (u + 0x7FFFu + ((u >> 16) & 1u)) >> 16;   // RNE
  unsigned short s = (unsigned short)r;
  return *reinterpret_cast<__bf16*>(&s);
}
__device__ inline float bf2f(__bf16 b) {
  unsigned short s = *reinterpret_cast<unsigned short*>(&b);
  return __uint_as_float(((unsigned)s) << 16);
}

__device__ inline void atomicMaxF(float* addr, float v) {
  if (v >= 0.0f) atomicMax((int*)addr, __float_as_int(v));
  else           atomicMin((unsigned int*)addr, __float_as_uint(v));
}

// ---- WMMA fragment helpers (CDNA5 16-bit A(16x32) / B(32x16) layouts) -----
__device__ inline v16bf load_a_frag(const __bf16* __restrict__ Arow, int k0, int hi) {
  // elems 0..7 -> K = k0+hi*8+e ; elems 8..15 -> K = k0+16+hi*8+e
  v8bf a0 = *(const v8bf*)(Arow + k0 + hi * 8);
  v8bf a1 = *(const v8bf*)(Arow + k0 + 16 + hi * 8);
  return __builtin_shufflevector(a0, a1, 0,1,2,3,4,5,6,7,8,9,10,11,12,13,14,15);
}
__device__ inline v16bf load_b_frag(const __bf16* __restrict__ Brow, int k0, int hi) {
  // elems 0..7 -> K = k0+hi*8+e ; elems 8..15 -> K = k0+16+hi*8+e  (column = lane&15 of B^T row)
  v8bf b0 = *(const v8bf*)(Brow + k0 + hi * 8);
  v8bf b1 = *(const v8bf*)(Brow + k0 + 16 + hi * 8);
  return __builtin_shufflevector(b0, b1, 0,1,2,3,4,5,6,7,8,9,10,11,12,13,14,15);
}
__device__ inline v8f wmma_bf16(v16bf a, v16bf b, v8f c) {
  return __builtin_amdgcn_wmma_f32_16x16x32_bf16(false, a, false, b, (short)0, c, false, false);
}

__device__ inline void store_tile(v8f acc, const float* __restrict__ bias,
                                  float* __restrict__ Cf, __bf16* __restrict__ Cb,
                                  int row0, int col, int N, int act, int hi) {
  float bb = bias ? bias[col] : 0.0f;
#pragma unroll
  for (int i = 0; i < 8; ++i) {
    float v = acc[i] + bb;
    if (act) v = v > 0.f ? v : 0.f;
    size_t idx = (size_t)(row0 + hi * 8 + i) * N + col;
    if (Cf) Cf[idx] = v;
    if (Cb) Cb[idx] = f2bf(v);
  }
}

// ---- generic bf16 WMMA GEMM:  C[M,N] = act(A[M,K] @ BT[N,K]^T + bias) -----
// One wave per 32x32 macro-tile (2x2 register blocking). Requirements met by
// every shape in this network: M%16==0, N%32==0, K%32==0. The last M block
// may be 16 rows tall (M/16 odd) -> wave-uniform half-height path.
__global__ void gemm_bf16_wmma(const __bf16* __restrict__ A,
                               const __bf16* __restrict__ BT,
                               const float* __restrict__ bias,
                               float* __restrict__ Cf,
                               __bf16* __restrict__ Cb,
                               int M, int N, int K, int act) {
  int wave = blockIdx.x * (blockDim.x >> 5) + (threadIdx.x >> 5);
  int mt = M >> 4;            // 16-row tiles
  int nb = N >> 5;            // 32-col blocks
  int mb = (mt + 1) >> 1;     // 32-row blocks (last may be half height)
  if (wave >= mb * nb) return;              // wave-uniform
  int bm = wave / nb, bn = wave % nb;
  int lane = threadIdx.x & 31;
  int r = lane & 15, hi = lane >> 4;
  bool rows2 = (bm * 2 + 2) <= mt;          // wave-uniform

  const __bf16* Arow0 = A  + (size_t)(bm * 32 + r) * K;
  const __bf16* Arow1 = Arow0 + (size_t)16 * K;
  const __bf16* Brow0 = BT + (size_t)(bn * 32 + r) * K;
  const __bf16* Brow1 = Brow0 + (size_t)16 * K;

  v8f acc00 = {0.f,0.f,0.f,0.f,0.f,0.f,0.f,0.f};
  v8f acc01 = acc00, acc10 = acc00, acc11 = acc00;

  if (rows2) {
    for (int k0 = 0; k0 < K; k0 += 32) {
      v16bf a0 = load_a_frag(Arow0, k0, hi);
      v16bf a1 = load_a_frag(Arow1, k0, hi);
      v16bf b0 = load_b_frag(Brow0, k0, hi);
      v16bf b1 = load_b_frag(Brow1, k0, hi);
      acc00 = wmma_bf16(a0, b0, acc00);
      acc01 = wmma_bf16(a0, b1, acc01);
      acc10 = wmma_bf16(a1, b0, acc10);
      acc11 = wmma_bf16(a1, b1, acc11);
    }
  } else {
    for (int k0 = 0; k0 < K; k0 += 32) {
      v16bf a0 = load_a_frag(Arow0, k0, hi);
      v16bf b0 = load_b_frag(Brow0, k0, hi);
      v16bf b1 = load_b_frag(Brow1, k0, hi);
      acc00 = wmma_bf16(a0, b0, acc00);
      acc01 = wmma_bf16(a0, b1, acc01);
    }
  }

  // C/D layout: VGPR i -> row = row0 + hi*8 + i, col = col0 + (lane&15)
  int row0 = bm * 32;
  int col0 = bn * 32 + r;
  store_tile(acc00, bias, Cf, Cb, row0, col0,      N, act, hi);
  store_tile(acc01, bias, Cf, Cb, row0, col0 + 16, N, act, hi);
  if (rows2) {
    store_tile(acc10, bias, Cf, Cb, row0 + 16, col0,      N, act, hi);
    store_tile(acc11, bias, Cf, Cb, row0 + 16, col0 + 16, N, act, hi);
  }
}

// ---- small utility kernels ------------------------------------------------
__global__ void k_fill_f32(float* p, float v, int n) {
  int i = blockIdx.x * blockDim.x + threadIdx.x;
  if (i < n) p[i] = v;
}
__global__ void k_f32_to_bf16(const float* __restrict__ in, __bf16* __restrict__ out, int n) {
  int i = blockIdx.x * blockDim.x + threadIdx.x;
  if (i < n) out[i] = f2bf(in[i]);
}
// in: f32 [K,N] row-major  ->  out: bf16 [N,K] row-major (i.e. B^T)
__global__ void k_transpose_bf16(const float* __restrict__ in, __bf16* __restrict__ out,
                                 int K, int N) {
  int i = blockIdx.x * blockDim.x + threadIdx.x;
  if (i >= K * N) return;
  int k = i / N, n = i - k * N;
  out[(size_t)n * K + k] = f2bf(in[i]);
}

// ---- edge pass 1: logits + scatter-max (one wave per edge) ----------------
__global__ void k_edge_logits(const float* __restrict__ xl, const float* __restrict__ xr,
                              const float* __restrict__ att,
                              const int* __restrict__ src, const int* __restrict__ dst,
                              float* __restrict__ elog, float* __restrict__ lmax) {
  int wid = blockIdx.x * (blockDim.x >> 5) + (threadIdx.x >> 5);
  if (wid >= NE) return;
  int lane = threadIdx.x & 31;
  int s = src[wid], d = dst[wid];
  const float* xls = xl + (size_t)s * HD;
  const float* xrd = xr + (size_t)d * HD;
#pragma unroll
  for (int h = 0; h < NH; ++h) {
    int c = h * 32 + lane;
    float p = xls[c] + xrd[c];
    p = p > 0.f ? p : 0.2f * p;                 // leaky_relu(msg + xr, 0.2)
    float v = p * att[c];
    for (int off = 16; off > 0; off >>= 1) v += __shfl_xor(v, off, 32);
    if (lane == 0) {
      elog[(size_t)wid * NH + h] = v;
      atomicMaxF(lmax + (size_t)d * NH + h, v);
    }
  }
}

// ---- edge pass 2: exp(logit - max), accumulate denominator ----------------
__global__ void k_edge_expsum(float* __restrict__ elog, const float* __restrict__ lmax,
                              float* __restrict__ denom, const int* __restrict__ dst) {
  int i = blockIdx.x * blockDim.x + threadIdx.x;
  if (i >= NE * NH) return;
  int e = i >> 3, h = i & 7;
  int d = dst[e];
  float ex = __expf(elog[i] - lmax[(size_t)d * NH + h]);
  elog[i] = ex;
  atomicAdd(denom + (size_t)d * NH + h, ex);
}

// ---- edge pass 3: alpha-weighted scatter of raw messages ------------------
__global__ void k_edge_scatter(const float* __restrict__ xl, const float* __restrict__ elog,
                               const float* __restrict__ denom,
                               const int* __restrict__ src, const int* __restrict__ dst,
                               float* __restrict__ acc) {
  int wid = blockIdx.x * (blockDim.x >> 5) + (threadIdx.x >> 5);
  if (wid >= NE) return;
  int lane = threadIdx.x & 31;
  int s = src[wid], d = dst[wid];
#pragma unroll
  for (int h = 0; h < NH; ++h) {
    float alpha = elog[(size_t)wid * NH + h] / (denom[(size_t)d * NH + h] + 1e-16f);
    int c = h * 32 + lane;
    atomicAdd(acc + (size_t)d * HD + c, xl[(size_t)s * HD + c] * alpha);
  }
}

// ---- hetero mean + relu + layernorm -> bf16 (one wave per node row) -------
__global__ void k_post_hetero(const float* __restrict__ acc,
                              const float* __restrict__ b1, const float* __restrict__ b2,
                              int N, __bf16* __restrict__ xout) {
  int row = blockIdx.x * (blockDim.x >> 5) + (threadIdx.x >> 5);
  if (row >= N) return;
  int lane = threadIdx.x & 31;
  float v[8];
  float sum = 0.f;
#pragma unroll
  for (int j = 0; j < 8; ++j) {
    int c = j * 32 + lane;
    float t = (acc[(size_t)row * HD + c] + b1[c] + b2[c]) * 0.5f;
    t = t > 0.f ? t : 0.f;
    v[j] = t; sum += t;
  }
  for (int off = 16; off > 0; off >>= 1) sum += __shfl_xor(sum, off, 32);
  float mean = sum * (1.0f / HD);
  float var = 0.f;
#pragma unroll
  for (int j = 0; j < 8; ++j) { float dd = v[j] - mean; var += dd * dd; }
  for (int off = 16; off > 0; off >>= 1) var += __shfl_xor(var, off, 32);
  float inv = rsqrtf(var * (1.0f / HD) + 1e-5f);
#pragma unroll
  for (int j = 0; j < 8; ++j)
    xout[(size_t)row * HD + j * 32 + lane] = f2bf((v[j] - mean) * inv);
}

// ---- final predictor head: [N,64] @ [64,1] + b ----------------------------
__global__ void k_perf_head(const __bf16* __restrict__ h, const float* __restrict__ w,
                            const float* __restrict__ b, float* __restrict__ out, int N) {
  int i = blockIdx.x * blockDim.x + threadIdx.x;
  if (i >= N) return;
  float s = 0.f;
#pragma unroll
  for (int j = 0; j < 64; ++j) s += bf2f(h[(size_t)i * 64 + j]) * w[j];
  out[i] = s + b[0];
}

// ---------------------------------------------------------------------------
extern "C" void kernel_launch(void* const* d_in, const int* in_sizes, int n_in,
                              void* d_out, int out_size, void* d_ws, size_t ws_size,
                              hipStream_t stream) {
  const float* x_player = (const float*)d_in[0];
  const float* x_venue  = (const float*)d_in[1];
  const int*   edges[4] = {(const int*)d_in[2], (const int*)d_in[3],
                           (const int*)d_in[4], (const int*)d_in[5]};
  const float* Wp_in = (const float*)d_in[6];  const float* bp_in = (const float*)d_in[7];
  const float* Wv_in = (const float*)d_in[8];  const float* bv_in = (const float*)d_in[9];
  const float* gat_Wl  = (const float*)d_in[10];
  const float* gat_Wr  = (const float*)d_in[11];
  const float* gat_att = (const float*)d_in[12];
  const float* gat_b   = (const float*)d_in[13];
  const float* out_W1 = (const float*)d_in[14]; const float* out_b1 = (const float*)d_in[15];
  const float* out_W2 = (const float*)d_in[16]; const float* out_b2 = (const float*)d_in[17];
  const float* pW1 = (const float*)d_in[18]; const float* pb1 = (const float*)d_in[19];
  const float* pW2 = (const float*)d_in[20]; const float* pb2 = (const float*)d_in[21];
  const float* pW3 = (const float*)d_in[22]; const float* pb3 = (const float*)d_in[23];

  float* outF   = (float*)d_out;
  float* embsPf = outF;                       // [NP,128]
  float* embsVf = outF + (size_t)NP * 128;    // [NV,128]
  float* perfF  = embsVf + (size_t)NV * 128;  // [NP]

  // ---- workspace carve ----------------------------------------------------
  size_t off = 0;
  auto alloc = [&](size_t bytes) -> void* {
    void* p = (void*)((char*)d_ws + off);
    off += (bytes + 255) & ~(size_t)255;
    return p;
  };
  __bf16* xpin  = (__bf16*)alloc((size_t)NP * 64 * 2);
  __bf16* xvin  = (__bf16*)alloc((size_t)NV * 32 * 2);
  __bf16* WpinT = (__bf16*)alloc(256 * 64 * 2);
  __bf16* WvinT = (__bf16*)alloc(256 * 32 * 2);
  __bf16* WlT   = (__bf16*)alloc((size_t)LYR * 4 * HD * HD * 2);
  __bf16* WrT   = (__bf16*)alloc((size_t)LYR * 4 * HD * HD * 2);
  __bf16* W1oT  = (__bf16*)alloc((size_t)2 * 128 * 256 * 2);
  __bf16* W2oT  = (__bf16*)alloc((size_t)2 * 128 * 128 * 2);
  __bf16* pW1T  = (__bf16*)alloc(128 * 128 * 2);
  __bf16* pW2T  = (__bf16*)alloc(64 * 128 * 2);
  __bf16* xsP   = (__bf16*)alloc((size_t)NP * HD * 2);
  __bf16* xsV   = (__bf16*)alloc((size_t)NV * HD * 2);
  __bf16* hbuf1 = (__bf16*)alloc((size_t)NP * 128 * 2);
  __bf16* hbuf2 = (__bf16*)alloc((size_t)NP * 128 * 2);
  __bf16* embsPb= (__bf16*)alloc((size_t)NP * 128 * 2);
  float*  xlb   = (float*)alloc((size_t)NP * HD * 4);
  float*  xrb   = (float*)alloc((size_t)NP * HD * 4);
  float*  accP  = (float*)alloc((size_t)NP * HD * 4);
  float*  accV  = (float*)alloc((size_t)NV * HD * 4);
  float*  lmax  = (float*)alloc((size_t)NP * NH * 4);
  float*  denom = (float*)alloc((size_t)NP * NH * 4);
  float*  elog  = (float*)alloc((size_t)NE * NH * 4);

  auto gemm = [&](const __bf16* A, const __bf16* BT, const float* bias,
                  float* Cf, __bf16* Cb, int M, int N, int K, int act) {
    int mb = ((M >> 4) + 1) >> 1;      // 32-row macro blocks
    int nb = N >> 5;                   // 32-col macro blocks
    int tiles = mb * nb;
    gemm_bf16_wmma<<<(tiles + 3) / 4, 128, 0, stream>>>(A, BT, bias, Cf, Cb, M, N, K, act);
  };
  auto transp = [&](const float* in, __bf16* out, int K, int N) {
    int n = K * N;
    k_transpose_bf16<<<(n + 255) / 256, 256, 0, stream>>>(in, out, K, N);
  };
  auto fill = [&](float* p, float v, int n) {
    k_fill_f32<<<(n + 255) / 256, 256, 0, stream>>>(p, v, n);
  };

  // ---- 1) weights -> transposed bf16 --------------------------------------
  transp(Wp_in, WpinT, 64, 256);
  transp(Wv_in, WvinT, 32, 256);
  for (int i = 0; i < LYR * 4; ++i) {
    transp(gat_Wl + (size_t)i * HD * HD, WlT + (size_t)i * HD * HD, HD, HD);
    transp(gat_Wr + (size_t)i * HD * HD, WrT + (size_t)i * HD * HD, HD, HD);
  }
  for (int t = 0; t < 2; ++t) {
    transp(out_W1 + (size_t)t * 256 * 128, W1oT + (size_t)t * 128 * 256, 256, 128);
    transp(out_W2 + (size_t)t * 128 * 128, W2oT + (size_t)t * 128 * 128, 128, 128);
  }
  transp(pW1, pW1T, 128, 128);   // ctx == 0, so only first 128 rows of pW1 matter
  transp(pW2, pW2T, 128, 64);

  // ---- 2) input features -> bf16, input projections -----------------------
  k_f32_to_bf16<<<(NP * 64 + 255) / 256, 256, 0, stream>>>(x_player, xpin, NP * 64);
  k_f32_to_bf16<<<(NV * 32 + 255) / 256, 256, 0, stream>>>(x_venue, xvin, NV * 32);
  gemm(xpin, WpinT, bp_in, nullptr, xsP, NP, 256, 64, 0);
  gemm(xvin, WvinT, bv_in, nullptr, xsV, NV, 256, 32, 0);

  // ---- 3) GATv2 layers -----------------------------------------------------
  const int etS[4] = {0, 0, 1, 1};   // src type per edge type (pp, pv, vp, vv)
  const int etD[4] = {0, 1, 0, 1};   // dst type
  const int Ns[2]  = {NP, NV};
  __bf16* xs[2] = {xsP, xsV};
  float* accs[2] = {accP, accV};

  for (int l = 0; l < LYR; ++l) {
    fill(accP, 0.0f, NP * HD);
    fill(accV, 0.0f, NV * HD);
    for (int k = 0; k < 4; ++k) {
      int s = etS[k], d = etD[k];
      const __bf16* wl = WlT + (size_t)(l * 4 + k) * HD * HD;
      const __bf16* wr = WrT + (size_t)(l * 4 + k) * HD * HD;
      gemm(xs[s], wl, nullptr, xlb, nullptr, Ns[s], HD, HD, 0);   // xl = x_src @ Wl
      gemm(xs[d], wr, nullptr, xrb, nullptr, Ns[d], HD, HD, 0);   // xr = x_dst @ Wr
      fill(lmax, -FLT_MAX, Ns[d] * NH);
      fill(denom, 0.0f, Ns[d] * NH);
      const int* src = edges[k];
      const int* dst = edges[k] + NE;
      const float* att = gat_att + (size_t)(l * 4 + k) * NH * 32;
      k_edge_logits<<<(NE + 7) / 8, 256, 0, stream>>>(xlb, xrb, att, src, dst, elog, lmax);
      k_edge_expsum<<<(NE * NH + 255) / 256, 256, 0, stream>>>(elog, lmax, denom, dst);
      k_edge_scatter<<<(NE + 7) / 8, 256, 0, stream>>>(xlb, elog, denom, src, dst, accs[d]);
    }
    // dst=p gets edge types 0,2 ; dst=v gets edge types 1,3 (mean of 2 each)
    k_post_hetero<<<(NP + 7) / 8, 256, 0, stream>>>(
        accP, gat_b + (size_t)(l * 4 + 0) * HD, gat_b + (size_t)(l * 4 + 2) * HD, NP, xsP);
    k_post_hetero<<<(NV + 7) / 8, 256, 0, stream>>>(
        accV, gat_b + (size_t)(l * 4 + 1) * HD, gat_b + (size_t)(l * 4 + 3) * HD, NV, xsV);
  }

  // ---- 4) output projections ----------------------------------------------
  gemm(xsP, W1oT, out_b1, nullptr, hbuf1, NP, 128, 256, 1);                     // relu
  gemm(hbuf1, W2oT, out_b2, embsPf, embsPb, NP, 128, 128, 0);                   // embs_p
  gemm(xsV, W1oT + (size_t)128 * 256, out_b1 + 128, nullptr, hbuf2, NV, 128, 256, 1);
  gemm(hbuf2, W2oT + (size_t)128 * 128, out_b2 + 128, embsVf, nullptr, NV, 128, 128, 0);

  // ---- 5) performance predictor (ctx == 0) --------------------------------
  gemm(embsPb, pW1T, pb1, nullptr, hbuf1, NP, 128, 128, 1);
  gemm(hbuf1, pW2T, pb2, nullptr, hbuf2, NP, 64, 128, 1);
  k_perf_head<<<(NP + 255) / 256, 256, 0, stream>>>(hbuf2, pW3, pb3, perfF, NP);

  (void)in_sizes; (void)n_in; (void)out_size; (void)ws_size;
}
